// SelfAttentionDecoder_60473139527978
// MI455X (gfx1250) — matrix operand order repaired and checked by reading, not compiled, not run on hardware
//
#include <hip/hip_runtime.h>
#include <hip/hip_bf16.h>

// ---------------------------------------------------------------------------
// Self-attention decoder for MI455X (gfx1250), bf16 WMMA path.
//   B=32, T=768, C=384, H=6, D=64.
// - All matmuls: V_WMMA_F32_16X16X32_BF16 (f32 accumulate).
// - Attention: K/V tiles staged in LDS per block via the Tensor Data Mover
//   (tensor_load_to_lds + s_wait_tensorcnt) when available, cooperative
//   loads otherwise; shared by all 4 waves of the block.
// ---------------------------------------------------------------------------

#define EMBED 384
#define NHEAD 6
#define HDIM  64
#define BB    32
#define TT    768
#define NTOK  (BB * TT)   // 24576

// LDS layout for the attention kernel (dynamic shared memory, base offset 0):
//   K tile : 64 rows (keys)  x 64 bf16, padded row stride 72 bf16 (144 B)
//   V tile : 64 rows (dims)  x 64 bf16, padded row stride 72 bf16
//   P stage: 4 waves x 16x32 bf16
#define KLD       72
#define LDS_K_OFF 0
#define LDS_V_OFF (64 * KLD * 2)            //  9216
#define LDS_P_OFF (2 * 64 * KLD * 2)        // 18432
#define LDS_TOTAL (LDS_P_OFF + 4 * 16 * 32 * 2)  // 22528 bytes

typedef __bf16 v16bf __attribute__((ext_vector_type(16)));
typedef __bf16 v8bf  __attribute__((ext_vector_type(8)));
typedef float  v8f   __attribute__((ext_vector_type(8)));
typedef unsigned int u32x4 __attribute__((ext_vector_type(4)));
typedef int          i32x8 __attribute__((ext_vector_type(8)));
typedef int          i32x4 __attribute__((ext_vector_type(4)));

#if __has_builtin(__builtin_amdgcn_tensor_load_to_lds)
#define HAVE_TDM 1
#endif

// D = A*B + C, bf16 inputs, f32 accumulate.
__device__ __forceinline__ v8f wmma_bf16(v16bf a, v16bf b, v8f c) {
    return __builtin_amdgcn_wmma_f32_16x16x32_bf16(false, a, false, b,
                                                   (short)0, c, false, false);
}

// A-fragment (16x32, M x K), row-major source with leading dim ld.
// Contiguous k order (same as B): lane<16 holds row M=lane, K 0..15;
//               lane>=16 holds row M=lane-16, K 16..31.
__device__ __forceinline__ v16bf load_a_frag(const __bf16* base, int ld,
                                             int row0, int col0, int lane) {
    int ln = lane & 15, half = lane >> 4;
    const __bf16* p = base + (size_t)(row0 + ln) * ld + col0 + half * 16;
    v8bf lo = *(const v8bf*)p;          // K = half*16 + 0..7
    v8bf hi = *(const v8bf*)(p + 8);    // K = half*16 + 8..15
    return __builtin_shufflevector(lo, hi, 0, 1, 2, 3, 4, 5, 6, 7,
                                   8, 9, 10, 11, 12, 13, 14, 15);
}

// B-fragment (32x16, K x N) where B[k][n] = base[(col0+n)*ld + k0 + k].
// Lanes 0..15 -> N=lane, K 0..15; lanes 16..31 -> N=lane-16, K 16..31.
// Global version: one 32-byte vector load (compiler splits into 2x b128).
__device__ __forceinline__ v16bf load_b_frag(const __bf16* base, int ld,
                                             int col0, int k0, int lane) {
    int ln = lane & 15, half = lane >> 4;
    return *(const v16bf*)(base + (size_t)(col0 + ln) * ld + k0 + half * 16);
}

// Same B-fragment from 16-byte-aligned (possibly padded-stride) storage: two
// v8bf loads so only 16-B alignment is assumed (LDS rows use stride 144 B).
__device__ __forceinline__ v16bf load_b_frag16(const __bf16* base, int ld,
                                               int col0, int k0, int lane) {
    int ln = lane & 15, half = lane >> 4;
    const __bf16* p = base + (size_t)(col0 + ln) * ld + k0 + half * 16;
    v8bf lo = *(const v8bf*)p;
    v8bf hi = *(const v8bf*)(p + 8);
    return __builtin_shufflevector(lo, hi, 0, 1, 2, 3, 4, 5, 6, 7,
                                   8, 9, 10, 11, 12, 13, 14, 15);
}

#ifdef HAVE_TDM
// Build a 2-D TDM descriptor: tile 64 rows x 64 bf16 -> LDS with a 4-DWORD pad
// after every 32 DWORDs (row stride in LDS becomes 72 bf16 = KLD).
__device__ __forceinline__ void tdm_tile_load(const void* gaddr, unsigned lds_off,
                                              unsigned row_stride_elems) {
    unsigned long long ga = (unsigned long long)(uintptr_t)gaddr;
    u32x4 g0;
    g0[0] = 1u;                                       // count=1 (valid D#)
    g0[1] = lds_off;                                  // lds_addr (bytes)
    g0[2] = (unsigned)(ga & 0xffffffffu);             // global_addr[31:0]
    g0[3] = (unsigned)((ga >> 32) & 0x1ffffffu)       // global_addr[56:32]
            | 0x80000000u;                            // type=2 ("image")
    i32x8 g1;
    g1[0] = (1 << 16)                                 // data_size = 2 bytes
          | (1 << 20)                                 // pad_enable
          | (4 << 22)                                 // pad_interval: 32 DWORDs
          | (3 << 25);                                // pad_amount: 4 DWORDs
    g1[1] = (int)(64u << 16);                         // tensor_dim0 = 64
    g1[2] = (int)(768u << 16);                        // tensor_dim1 = 768
    g1[3] = (int)(64u << 16);                         // tile_dim0 = 64
    g1[4] = 64;                                       // tile_dim1 = 64
    g1[5] = (int)row_stride_elems;                    // tensor_dim0_stride
    g1[6] = 0;
    g1[7] = 0;
    i32x4 z4 = {};
#if __clang_major__ >= 23
    i32x8 z8 = {};
    __builtin_amdgcn_tensor_load_to_lds(g0, g1, z4, z4, z8, 0);
#else
    __builtin_amdgcn_tensor_load_to_lds(g0, g1, z4, z4, 0);
#endif
}
#endif

__device__ __forceinline__ void wait_tensorcnt0() {
#if __has_builtin(__builtin_amdgcn_s_wait_tensorcnt)
    __builtin_amdgcn_s_wait_tensorcnt((short)0);
#else
    asm volatile("s_wait_tensorcnt 0" ::: "memory");
#endif
}

// ---------------------------------------------------------------------------
__global__ __launch_bounds__(256) void cvt_f32_bf16(const float* __restrict__ src,
                                                    __bf16* __restrict__ dst, int n) {
    int i = blockIdx.x * 256 + threadIdx.x;
    if (i < n) dst[i] = (__bf16)src[i];
}

// ---------------------------------------------------------------------------
// QKV projection: Y = X @ W^T  (W is [out,in], torch convention).
// blockIdx.z selects Q/K/V. Q,K stored [B,H,T,D]; V stored transposed [B,H,D,T].
__global__ __launch_bounds__(128) void qkv_kernel(
    const __bf16* __restrict__ X,
    const __bf16* __restrict__ Wq, const __bf16* __restrict__ Wk,
    const __bf16* __restrict__ Wv,
    __bf16* __restrict__ Q, __bf16* __restrict__ K, __bf16* __restrict__ Vt) {
    int wave = threadIdx.x >> 5, lane = threadIdx.x & 31;
    int ln = lane & 15, half = lane >> 4;
    int rows = blockIdx.x * 64 + wave * 16;   // token-tile base
    int cols = blockIdx.y * 64;               // output-feature tile base
    int which = blockIdx.z;
    const __bf16* W = (which == 0) ? Wq : (which == 1) ? Wk : Wv;

    v8f acc[4] = {};
    for (int kc = 0; kc < EMBED; kc += 32) {
        if (kc + 32 < EMBED)                  // gfx1250 global_prefetch_b8
            __builtin_prefetch(X + (size_t)(rows + ln) * EMBED + kc + 32, 0, 1);
        v16bf a = load_a_frag(X, EMBED, rows, kc, lane);
#pragma unroll
        for (int ct = 0; ct < 4; ct++) {
            v16bf b = load_b_frag(W, EMBED, cols + ct * 16, kc, lane);
            acc[ct] = wmma_bf16(a, b, acc[ct]);
        }
    }
#pragma unroll
    for (int ct = 0; ct < 4; ct++) {
#pragma unroll
        for (int r = 0; r < 8; r++) {
            int rg = rows + r + half * 8;     // global token index
            int og = cols + ct * 16 + ln;     // global output feature
            int bI = rg / TT, t = rg % TT;
            int hI = og >> 6, d = og & 63;
            __bf16 v = (__bf16)acc[ct][r];
            if (which == 2)
                Vt[(((size_t)bI * NHEAD + hI) * HDIM + d) * TT + t] = v;
            else if (which == 0)
                Q[(((size_t)bI * NHEAD + hI) * TT + t) * HDIM + d] = v;
            else
                K[(((size_t)bI * NHEAD + hI) * TT + t) * HDIM + d] = v;
        }
    }
}

// ---------------------------------------------------------------------------
// Flash attention. Block = 4 waves covering 64 query rows of one (b,h).
// Per 64-key chunk: TDM (or cooperative) load of K/V tiles into LDS shared by
// all waves; each wave runs scores -> online softmax -> P.V on its 16 rows.
__global__ __launch_bounds__(128) void attn_kernel(
    const __bf16* __restrict__ Q, const __bf16* __restrict__ K,
    const __bf16* __restrict__ Vt, __bf16* __restrict__ Ob) {
    extern __shared__ __bf16 smem[];
    __bf16* Ksh = smem;                               // [64][KLD]
    __bf16* Vsh = smem + 64 * KLD;                    // [64][KLD]
    int wave = threadIdx.x >> 5, lane = threadIdx.x & 31;
    int ln = lane & 15, half = lane >> 4;
    __bf16* pst = smem + 2 * 64 * KLD + wave * (16 * 32);
    int b = blockIdx.z, h = blockIdx.y;
    int base = blockIdx.x * 64;
    int qb = base + wave * 16;                        // this wave's query tile

    const __bf16* Qbh = Q  + ((size_t)b * NHEAD + h) * TT * HDIM;
    const __bf16* Kbh = K  + ((size_t)b * NHEAD + h) * TT * HDIM;
    const __bf16* Vbh = Vt + ((size_t)b * NHEAD + h) * HDIM * TT;

    // Preload Q A-fragments (16 x 64 = 2 k-steps of 32).
    v16bf aQ[2];
#pragma unroll
    for (int ks = 0; ks < 2; ks++) aQ[ks] = load_a_frag(Qbh, HDIM, qb, ks * 32, lane);

    v8f acc[4] = {};                                  // 16 x 64 output accum, f32
    float rowm[8], rowl[8];
#pragma unroll
    for (int r = 0; r < 8; r++) { rowm[r] = -1e30f; rowl[r] = 0.f; }

    for (int sb = 0; sb < base + 64; sb += 64) {
        // ---- stage K tile [s 0..63][d 0..63] and V tile [d 0..63][s 0..63]
#ifdef HAVE_TDM
        if (wave == 0) {
            tdm_tile_load(Kbh + (size_t)sb * HDIM, LDS_K_OFF, HDIM);
            tdm_tile_load(Vbh + sb,                LDS_V_OFF, TT);
            wait_tensorcnt0();
        }
#else
#pragma unroll
        for (int i = 0; i < 4; i++) {                 // 512 segs of 16 B per tile
            int seg = threadIdx.x + i * 128;
            int row = seg >> 3, part = seg & 7;
            *(v8bf*)(Ksh + row * KLD + part * 8) =
                *(const v8bf*)(Kbh + (size_t)(sb + row) * HDIM + part * 8);
            *(v8bf*)(Vsh + row * KLD + part * 8) =
                *(const v8bf*)(Vbh + (size_t)row * TT + sb + part * 8);
        }
#endif
        __syncthreads();

#pragma unroll
        for (int sc = 0; sc < 2; sc++) {              // two 32-key sub-chunks
            int s0 = sb + sc * 32;
            if (s0 >= qb + 16) continue;              // fully above the diagonal
            // ---- scores: two 16x16 tiles, K-dim = d = 64
            v8f S0 = {}, S1 = {};
#pragma unroll
            for (int ks = 0; ks < 2; ks++) {
                v16bf b0 = load_b_frag16(Ksh, KLD, sc * 32,      ks * 32, lane);
                v16bf b1 = load_b_frag16(Ksh, KLD, sc * 32 + 16, ks * 32, lane);
                S0 = wmma_bf16(aQ[ks], b0, S0);
                S1 = wmma_bf16(aQ[ks], b1, S1);
            }
            // ---- mask + online softmax (row = r + half*8, col = ln in D layout)
#pragma unroll
            for (int r = 0; r < 8; r++) {
                int t = qb + r + half * 8;
                float v0 = S0[r] * 0.125f;            // HEAD_SIZE^-0.5
                float v1 = S1[r] * 0.125f;
                if (s0 + ln > t)      v0 = -1e30f;
                if (s0 + 16 + ln > t) v1 = -1e30f;
                float mx = fmaxf(v0, v1);
#pragma unroll
                for (int o = 8; o > 0; o >>= 1) mx = fmaxf(mx, __shfl_xor(mx, o, 32));
                float nm   = fmaxf(rowm[r], mx);
                float corr = __expf(rowm[r] - nm);
                float p0 = __expf(v0 - nm);
                float p1 = __expf(v1 - nm);
                float ps = p0 + p1;
#pragma unroll
                for (int o = 8; o > 0; o >>= 1) ps += __shfl_xor(ps, o, 32);
                rowl[r] = rowl[r] * corr + ps;
                rowm[r] = nm;
#pragma unroll
                for (int dt = 0; dt < 4; dt++) acc[dt][r] *= corr;
                int m = r + half * 8;                 // stage P (16x32) in LDS
                pst[m * 32 + ln]      = (__bf16)p0;
                pst[m * 32 + 16 + ln] = (__bf16)p1;
            }
            // Intra-wave D-layout -> A-layout round trip through LDS.
            asm volatile("s_wait_dscnt 0" ::: "memory");
            v16bf aP = load_a_frag(pst, 32, 0, 0, lane);
            asm volatile("" ::: "memory");
            // ---- O += P (16x32) x V (32x64), V tile is [d][s] in LDS
#pragma unroll
            for (int dt = 0; dt < 4; dt++) {
                v16bf bV = load_b_frag16(Vsh, KLD, dt * 16, sc * 32, lane);
                acc[dt] = wmma_bf16(aP, bV, acc[dt]);
            }
        }
        __syncthreads();                              // protect LDS reuse
    }
    // ---- epilogue: normalize, write [B*T, C] with head re-concat
#pragma unroll
    for (int dt = 0; dt < 4; dt++) {
#pragma unroll
        for (int r = 0; r < 8; r++) {
            int t = qb + r + half * 8;
            float v = acc[dt][r] / rowl[r];
            Ob[((size_t)b * TT + t) * EMBED + h * HDIM + dt * 16 + ln] = (__bf16)v;
        }
    }
}

// ---------------------------------------------------------------------------
// Output projection: out = Ob @ Wo^T + bo, f32 output.
__global__ __launch_bounds__(128) void oproj_kernel(
    const __bf16* __restrict__ Ob, const __bf16* __restrict__ Wo,
    const float* __restrict__ bo, float* __restrict__ out) {
    int wave = threadIdx.x >> 5, lane = threadIdx.x & 31;
    int ln = lane & 15, half = lane >> 4;
    int rows = blockIdx.x * 64 + wave * 16;
    int cols = blockIdx.y * 64;

    v8f acc[4] = {};
    for (int kc = 0; kc < EMBED; kc += 32) {
        if (kc + 32 < EMBED)
            __builtin_prefetch(Ob + (size_t)(rows + ln) * EMBED + kc + 32, 0, 1);
        v16bf a = load_a_frag(Ob, EMBED, rows, kc, lane);
#pragma unroll
        for (int ct = 0; ct < 4; ct++) {
            v16bf b = load_b_frag(Wo, EMBED, cols + ct * 16, kc, lane);
            acc[ct] = wmma_bf16(a, b, acc[ct]);
        }
    }
#pragma unroll
    for (int ct = 0; ct < 4; ct++) {
        int og = cols + ct * 16 + ln;
        float bias = bo[og];
#pragma unroll
        for (int r = 0; r < 8; r++) {
            int rg = rows + r + half * 8;
            out[(size_t)rg * EMBED + og] = acc[ct][r] + bias;
        }
    }
}

// ---------------------------------------------------------------------------
extern "C" void kernel_launch(void* const* d_in, const int* in_sizes, int n_in,
                              void* d_out, int out_size, void* d_ws, size_t ws_size,
                              hipStream_t stream) {
    const float* X  = (const float*)d_in[0];
    const float* Wq = (const float*)d_in[1];
    const float* Wk = (const float*)d_in[2];
    const float* Wv = (const float*)d_in[3];
    const float* Wo = (const float*)d_in[4];
    const float* bo = (const float*)d_in[5];

    // Workspace layout (~96 MB): bf16 X, 4 bf16 weights, Q, K, V^T, attn-out.
    char* ws = (char*)d_ws;
    size_t off = 0;
    auto carve = [&](size_t bytes) {
        char* p = ws + off;
        off = (off + bytes + 255) & ~(size_t)255;
        return p;
    };
    const size_t nX = (size_t)NTOK * EMBED;           // 9,437,184
    const size_t nW = (size_t)EMBED * EMBED;          // 147,456
    __bf16* Xb  = (__bf16*)carve(nX * 2);
    __bf16* Wqb = (__bf16*)carve(nW * 2);
    __bf16* Wkb = (__bf16*)carve(nW * 2);
    __bf16* Wvb = (__bf16*)carve(nW * 2);
    __bf16* Wob = (__bf16*)carve(nW * 2);
    __bf16* Qb  = (__bf16*)carve(nX * 2);             // [B,H,T,D]
    __bf16* Kb  = (__bf16*)carve(nX * 2);             // [B,H,T,D]
    __bf16* Vtb = (__bf16*)carve(nX * 2);             // [B,H,D,T]
    __bf16* Obf = (__bf16*)carve(nX * 2);             // [B*T, C]

    cvt_f32_bf16<<<(int)((nX + 255) / 256), 256, 0, stream>>>(X, Xb, (int)nX);
    cvt_f32_bf16<<<(int)((nW + 255) / 256), 256, 0, stream>>>(Wq, Wqb, (int)nW);
    cvt_f32_bf16<<<(int)((nW + 255) / 256), 256, 0, stream>>>(Wk, Wkb, (int)nW);
    cvt_f32_bf16<<<(int)((nW + 255) / 256), 256, 0, stream>>>(Wv, Wvb, (int)nW);
    cvt_f32_bf16<<<(int)((nW + 255) / 256), 256, 0, stream>>>(Wo, Wob, (int)nW);

    qkv_kernel<<<dim3(NTOK / 64, EMBED / 64, 3), 128, 0, stream>>>(
        Xb, Wqb, Wkb, Wvb, Qb, Kb, Vtb);

    attn_kernel<<<dim3(TT / 64, NHEAD, BB), 128, LDS_TOTAL, stream>>>(
        Qb, Kb, Vtb, Obf);

    oproj_kernel<<<dim3(NTOK / 64, EMBED / 64), 128, 0, stream>>>(
        Obf, Wob, bo, (float*)d_out);
}
